// ModifiedViTSelfAttention_7292854469092
// MI455X (gfx1250) — matrix operand order repaired and checked by reading, not compiled, not run on hardware
//
#include <hip/hip_runtime.h>
#include <math.h>

// ---------------------------------------------------------------------------
// ViT self-attention forward (gamma-LRP forward == plain attention).
// B=4, S=2048, H=768, heads=12, d=64.
// Phase 1: Q = (X Wq^T + bq) * 0.125  -> bf16 [b,h,s,64]
//          K = (X Wk^T + bk)          -> bf16 [b,h,s,64]
//          VT = (Wv X^T + bv)         -> bf16 [b,h,64,s]   (V transposed)
//          (32x32 output tile per wave: 2x2 WMMA register blocking)
// Phase 2: flash-style online-softmax attention; K/V tiles staged in LDS
//          once per workgroup (8 waves share), all matmuls on
//          v_wmma_f32_16x16x32_bf16.
// Workspace: 3 * 4*12*2048*64 bf16 = 37,748,736 bytes.
// ---------------------------------------------------------------------------

typedef __attribute__((ext_vector_type(8)))  __bf16 v8bf;
typedef __attribute__((ext_vector_type(16))) __bf16 v16bf;
typedef __attribute__((ext_vector_type(8)))  float  v8f;

#define LOG2E 1.44269504088896340736f

static __device__ inline v16bf cat8(v8bf a, v8bf b) {
  return __builtin_shufflevector(a, b, 0,1,2,3,4,5,6,7,8,9,10,11,12,13,14,15);
}

static __device__ inline v8f wmma_bf16(v16bf a, v16bf b, v8f c) {
  return __builtin_amdgcn_wmma_f32_16x16x32_bf16(false, a, false, b, (short)0, c,
                                                 false, false);
}

// A-fragment (16x32 bf16, M x K) from f32 row-major; `row` = lane's row ptr.
// K(e) = 32c + 16g + e  -> one contiguous 16-float segment.
static __device__ inline v16bf a_frag_f32(const float* __restrict__ row, int c, int g) {
  int d0 = 32 * c + 16 * g;
  float4 f0 = *(const float4*)(row + d0);
  float4 f1 = *(const float4*)(row + d0 + 4);
  float4 f2 = *(const float4*)(row + d0 + 8);
  float4 f3 = *(const float4*)(row + d0 + 12);
  v16bf r;
  r[0]=(__bf16)f0.x; r[1]=(__bf16)f0.y; r[2]=(__bf16)f0.z; r[3]=(__bf16)f0.w;
  r[4]=(__bf16)f1.x; r[5]=(__bf16)f1.y; r[6]=(__bf16)f1.z; r[7]=(__bf16)f1.w;
  r[8]=(__bf16)f2.x; r[9]=(__bf16)f2.y; r[10]=(__bf16)f2.z; r[11]=(__bf16)f2.w;
  r[12]=(__bf16)f3.x; r[13]=(__bf16)f3.y; r[14]=(__bf16)f3.z; r[15]=(__bf16)f3.w;
  return r;
}

// B-fragment (32x16 bf16, K x N) from f32 row-major where B[k,n] = Mtx[n_row][k].
// K(e) = 32c + 16g + e  -> one contiguous 16-float segment.
static __device__ inline v16bf b_frag_f32(const float* __restrict__ row, int c, int g) {
  int d0 = 32 * c + 16 * g;
  float4 f0 = *(const float4*)(row + d0);
  float4 f1 = *(const float4*)(row + d0 + 4);
  float4 f2 = *(const float4*)(row + d0 + 8);
  float4 f3 = *(const float4*)(row + d0 + 12);
  v16bf r;
  r[0]=(__bf16)f0.x; r[1]=(__bf16)f0.y; r[2]=(__bf16)f0.z; r[3]=(__bf16)f0.w;
  r[4]=(__bf16)f1.x; r[5]=(__bf16)f1.y; r[6]=(__bf16)f1.z; r[7]=(__bf16)f1.w;
  r[8]=(__bf16)f2.x; r[9]=(__bf16)f2.y; r[10]=(__bf16)f2.z; r[11]=(__bf16)f2.w;
  r[12]=(__bf16)f3.x; r[13]=(__bf16)f3.y; r[14]=(__bf16)f3.z; r[15]=(__bf16)f3.w;
  return r;
}

// ---------------------------------------------------------------------------
// Phase 1a: Q/K projection, 32x32 tile per wave (2x2 register blocking).
// Y[row,o] = sum_d X[row,d]*W[o,d] + b[o]; bf16 out in [b,h,s,64] layout.
// grid.x = (8192/32) * (768/128) = 256 * 6 = 1536, 128 threads (4 waves).
// ---------------------------------------------------------------------------
__global__ __launch_bounds__(128) void qk_proj_kernel(
    const float* __restrict__ X, const float* __restrict__ W,
    const float* __restrict__ bias, __bf16* __restrict__ dst, float scale) {
  int wave = threadIdx.x >> 5, lane = threadIdx.x & 31;
  int g = lane >> 4, ln = lane & 15;
  int rowtile = blockIdx.x / 6, colgroup = blockIdx.x % 6;
  int rowbase = rowtile * 32;
  int obase = (colgroup * 4 + wave) * 32;

  const float* xr0 = X + (size_t)(rowbase + ln) * 768;
  const float* xr1 = xr0 + (size_t)16 * 768;
  const float* wr0 = W + (size_t)(obase + ln) * 768;
  const float* wr1 = wr0 + (size_t)16 * 768;

  v8f acc00 = {}, acc01 = {}, acc10 = {}, acc11 = {};
#pragma unroll
  for (int c = 0; c < 24; ++c) {
    v16bf a0 = a_frag_f32(xr0, c, g);
    v16bf a1 = a_frag_f32(xr1, c, g);
    v16bf b0 = b_frag_f32(wr0, c, g);
    v16bf b1 = b_frag_f32(wr1, c, g);
    acc00 = wmma_bf16(a0, b0, acc00);
    acc01 = wmma_bf16(a0, b1, acc01);
    acc10 = wmma_bf16(a1, b0, acc10);
    acc11 = wmma_bf16(a1, b1, acc11);
  }

#pragma unroll
  for (int j = 0; j < 2; ++j) {
    int o = obase + 16 * j + ln;       // output column for this lane
    float bo = bias[o];
    int h = o >> 6, d = o & 63;
#pragma unroll
    for (int i = 0; i < 2; ++i) {
      v8f acc = (i == 0) ? ((j == 0) ? acc00 : acc01)
                         : ((j == 0) ? acc10 : acc11);
#pragma unroll
      for (int r = 0; r < 8; ++r) {
        int row = rowbase + 16 * i + r + 8 * g;   // global seq row [0,8192)
        int bb = row >> 11, s = row & 2047;
        float val = (acc[r] + bo) * scale;
        dst[(((size_t)bb * 12 + h) * 2048 + s) * 64 + d] = (__bf16)val;
      }
    }
  }
}

// ---------------------------------------------------------------------------
// Phase 1b: V projection, transposed, 32x32 tile per wave.
// VT[o,row] = sum_d W[o,d]*X[row,d] + b[o]; bf16 out in [b,h,64,s] layout.
// grid.x = (768/32) * (8192/128) = 24 * 64 = 1536, 128 threads.
// ---------------------------------------------------------------------------
__global__ __launch_bounds__(128) void v_proj_kernel(
    const float* __restrict__ X, const float* __restrict__ W,
    const float* __restrict__ bias, __bf16* __restrict__ dst) {
  int wave = threadIdx.x >> 5, lane = threadIdx.x & 31;
  int g = lane >> 4, ln = lane & 15;
  int otile = blockIdx.x / 64, sgroup = blockIdx.x % 64;
  int obase = otile * 32;
  int sbase = (sgroup * 4 + wave) * 32;

  const float* wr0 = W + (size_t)(obase + ln) * 768;   // A rows = W rows (o)
  const float* wr1 = wr0 + (size_t)16 * 768;
  const float* xr0 = X + (size_t)(sbase + ln) * 768;   // B cols = X rows (s)
  const float* xr1 = xr0 + (size_t)16 * 768;

  v8f acc00 = {}, acc01 = {}, acc10 = {}, acc11 = {};
#pragma unroll
  for (int c = 0; c < 24; ++c) {
    v16bf a0 = a_frag_f32(wr0, c, g);
    v16bf a1 = a_frag_f32(wr1, c, g);
    v16bf b0 = b_frag_f32(xr0, c, g);
    v16bf b1 = b_frag_f32(xr1, c, g);
    acc00 = wmma_bf16(a0, b0, acc00);
    acc01 = wmma_bf16(a0, b1, acc01);
    acc10 = wmma_bf16(a1, b0, acc10);
    acc11 = wmma_bf16(a1, b1, acc11);
  }

#pragma unroll
  for (int j = 0; j < 2; ++j) {
    int srow = sbase + 16 * j + ln;      // global sequence row
    int bb = srow >> 11, s = srow & 2047;
#pragma unroll
    for (int i = 0; i < 2; ++i) {
      v8f acc = (i == 0) ? ((j == 0) ? acc00 : acc01)
                         : ((j == 0) ? acc10 : acc11);
#pragma unroll
      for (int r = 0; r < 8; ++r) {
        int o = obase + 16 * i + r + 8 * g;
        int h = o >> 6, d = o & 63;
        float val = acc[r] + bias[o];
        dst[(((size_t)bb * 12 + h) * 64 + d) * 2048 + s] = (__bf16)val;
      }
    }
  }
}

// ---------------------------------------------------------------------------
// Phase 2: attention. 256 threads = 8 waves; each wave owns a 16-row q tile
// (block covers 128 q rows of one (b,h)). Per 32-key block the whole
// workgroup stages K (32x64) and V^T (64x32) tiles in LDS once; all 8 waves
// build WMMA fragments from LDS (8x less global K/V traffic).
// grid.x = 4 * 12 * (2048/128) = 768.
// ---------------------------------------------------------------------------
__global__ __launch_bounds__(256) void attn_kernel(
    const __bf16* __restrict__ Qb, const __bf16* __restrict__ Kb,
    const __bf16* __restrict__ VTb, float* __restrict__ out) {
  __shared__ __align__(32) __bf16 ktile[32][64];     // 4 KB  [key][d]
  __shared__ __align__(32) __bf16 vtile[64][32];     // 4 KB  [d][key]
  __shared__ __align__(32) __bf16 plds[8][16][32];   // 8 KB  per-wave P

  int tid = threadIdx.x;
  int wave = tid >> 5, lane = tid & 31;
  int g = lane >> 4, ln = lane & 15;

  int bi = blockIdx.x;
  int bb = bi / 192;
  int rem = bi % 192;
  int h = rem / 16;
  int qt = rem % 16;
  int qbase = qt * 128 + wave * 16;

  const __bf16* Qp = Qb + (((size_t)bb * 12 + h) * 2048) * 64;
  const __bf16* Kp = Kb + (((size_t)bb * 12 + h) * 2048) * 64;
  const __bf16* Vp = VTb + (((size_t)bb * 12 + h) * 64) * 2048;

  // staging assignments (one 16-byte copy each per iteration)
  int kkey = tid >> 3, kseg = (tid & 7) * 8;   // K tile: 32 rows x 64 halfs
  int vrow = tid >> 2, vseg = (tid & 3) * 8;   // V tile: 64 rows x 32 halfs

  // Q A-fragments (already scaled by 1/8 in projection)
  const __bf16* qr = Qp + (size_t)(qbase + ln) * 64;
  v16bf qa0 = cat8(*(const v8bf*)(qr + 16 * g), *(const v8bf*)(qr + 8 + 16 * g));
  v16bf qa1 = cat8(*(const v8bf*)(qr + 32 + 16 * g), *(const v8bf*)(qr + 40 + 16 * g));

  v8f o0 = {}, o1 = {}, o2 = {}, o3 = {};
  float mrow[8], lrow[8];
#pragma unroll
  for (int r = 0; r < 8; ++r) { mrow[r] = -__builtin_inff(); lrow[r] = 0.f; }

  for (int kk = 0; kk < 2048; kk += 32) {
    __syncthreads();   // previous iteration's consumers are done with K/V LDS
    *(v8bf*)&ktile[kkey][kseg] =
        *(const v8bf*)(Kp + (size_t)(kk + kkey) * 64 + kseg);
    *(v8bf*)&vtile[vrow][vseg] =
        *(const v8bf*)(Vp + (size_t)vrow * 2048 + kk + vseg);
    if (kk + 32 < 2048) {  // prefetch next block (global_prefetch_b8)
      __builtin_prefetch(Kp + (size_t)(kk + 32 + kkey) * 64 + kseg, 0, 0);
      __builtin_prefetch(Vp + (size_t)vrow * 2048 + kk + 32 + vseg, 0, 0);
    }
    __syncthreads();

    // K B-fragments from LDS: keys [0..15] and [16..31], d chunks 0-31/32-63
    v16bf kb00 = *(const v16bf*)&ktile[ln][16 * g];
    v16bf kb01 = *(const v16bf*)&ktile[ln][32 + 16 * g];
    v16bf kb10 = *(const v16bf*)&ktile[16 + ln][16 * g];
    v16bf kb11 = *(const v16bf*)&ktile[16 + ln][32 + 16 * g];

    v8f s0 = {}, s1 = {};
    s0 = wmma_bf16(qa0, kb00, s0);
    s0 = wmma_bf16(qa1, kb01, s0);
    s1 = wmma_bf16(qa0, kb10, s1);
    s1 = wmma_bf16(qa1, kb11, s1);

    // online softmax per row; row m=r+8g lives entirely in one 16-lane group
#pragma unroll
    for (int r = 0; r < 8; ++r) {
      float v0 = s0[r], v1 = s1[r];
      float mx = fmaxf(v0, v1);
      mx = fmaxf(mx, __shfl_xor(mx, 1, 16));
      mx = fmaxf(mx, __shfl_xor(mx, 2, 16));
      mx = fmaxf(mx, __shfl_xor(mx, 4, 16));
      mx = fmaxf(mx, __shfl_xor(mx, 8, 16));
      float mn = fmaxf(mrow[r], mx);
      float alpha = exp2f((mrow[r] - mn) * LOG2E);
      float p0 = exp2f((v0 - mn) * LOG2E);
      float p1 = exp2f((v1 - mn) * LOG2E);
      float ps = p0 + p1;
      ps += __shfl_xor(ps, 1, 16);
      ps += __shfl_xor(ps, 2, 16);
      ps += __shfl_xor(ps, 4, 16);
      ps += __shfl_xor(ps, 8, 16);
      lrow[r] = lrow[r] * alpha + ps;
      mrow[r] = mn;
      o0[r] *= alpha; o1[r] *= alpha; o2[r] *= alpha; o3[r] *= alpha;
      int m = r + 8 * g;
      plds[wave][m][ln] = (__bf16)p0;        // keys kk..kk+15
      plds[wave][m][16 + ln] = (__bf16)p1;   // keys kk+16..kk+31
    }

    // reload P (16x32) as bf16 A-fragment (wave-private, DS ops in-order)
    const __bf16* pr = &plds[wave][ln][0];
    v16bf pa = cat8(*(const v8bf*)(pr + 16 * g), *(const v8bf*)(pr + 8 + 16 * g));

    // V B-fragments from LDS V^T tile: row = d, contiguous along keys
    o0 = wmma_bf16(pa, *(const v16bf*)&vtile[ln][16 * g], o0);
    o1 = wmma_bf16(pa, *(const v16bf*)&vtile[16 + ln][16 * g], o1);
    o2 = wmma_bf16(pa, *(const v16bf*)&vtile[32 + ln][16 * g], o2);
    o3 = wmma_bf16(pa, *(const v16bf*)&vtile[48 + ln][16 * g], o3);
  }

  // epilogue: divide by softmax denominator, write [b, s, h*64+d] f32
  int hbase = h * 64;
#pragma unroll
  for (int r = 0; r < 8; ++r) {
    int srow = qbase + r + 8 * g;
    float inv = 1.f / lrow[r];
    size_t base = ((size_t)bb * 2048 + srow) * 768 + hbase + ln;
    out[base] = o0[r] * inv;
    out[base + 16] = o1[r] * inv;
    out[base + 32] = o2[r] * inv;
    out[base + 48] = o3[r] * inv;
  }
}

// ---------------------------------------------------------------------------
extern "C" void kernel_launch(void* const* d_in, const int* in_sizes, int n_in,
                              void* d_out, int out_size, void* d_ws, size_t ws_size,
                              hipStream_t stream) {
  const float* X  = (const float*)d_in[0];
  const float* Wq = (const float*)d_in[1];
  const float* bq = (const float*)d_in[2];
  const float* Wk = (const float*)d_in[3];
  const float* bk = (const float*)d_in[4];
  const float* Wv = (const float*)d_in[5];
  const float* bv = (const float*)d_in[6];
  float* out = (float*)d_out;

  const size_t QKV_ELEMS = (size_t)4 * 12 * 2048 * 64;  // 6,291,456 per tensor
  __bf16* Qbuf = (__bf16*)d_ws;
  __bf16* Kbuf = Qbuf + QKV_ELEMS;
  __bf16* VTbuf = Kbuf + QKV_ELEMS;
  // requires ws_size >= 3 * QKV_ELEMS * 2 = 37,748,736 bytes

  // Q projection pre-scaled by 1/sqrt(64) = 0.125 (folds softmax scale)
  qk_proj_kernel<<<1536, 128, 0, stream>>>(X, Wq, bq, Qbuf, 0.125f);
  qk_proj_kernel<<<1536, 128, 0, stream>>>(X, Wk, bk, Kbuf, 1.0f);
  v_proj_kernel<<<1536, 128, 0, stream>>>(X, Wv, bv, VTbuf);
  attn_kernel<<<768, 256, 0, stream>>>(Qbuf, Kbuf, VTbuf, out);
}
